// PNA_68401649156285
// MI455X (gfx1250) — matrix operand order repaired and checked by reading, not compiled, not run on hardware
//
#include <hip/hip_runtime.h>
#include <hip/hip_bf16.h>
#include <math.h>

// ---- problem constants (match reference) ----
#define NNODES  32768
#define NEDGES  262144
#define BGRAPH  64
#define NPG     512
#define TOPK    51
#define DD      75
#define EDD     50
#define TT      5
#define FOUT_   15
#define TD      375      // TT*DD
#define PRE_K   200
#define PRE_KP  224      // padded K for pre GEMM (7*32)
#define PRE_NC  7        // K chunks
#define PRE_NT  24       // N tiles (384 >= 375)
#define POST_K  975
#define POST_KP 992      // 31*32
#define POST_NC 31
#define LIN_NC  3        // 96 >= 75
#define LIN_KP  96
#define LIN_NT  5        // 80 >= 75
#define EPS_    1e-5f

typedef __attribute__((ext_vector_type(16))) _Float16 v16h;
typedef __attribute__((ext_vector_type(8)))  _Float16 v8h;
typedef __attribute__((ext_vector_type(8)))  float    v8f;

// monotone float <-> uint32 key (order-preserving) for single-instruction
// atomic min/max on floats
__device__ __forceinline__ unsigned f2ord(float v) {
  unsigned u = __float_as_uint(v);
  return (u & 0x80000000u) ? ~u : (u | 0x80000000u);
}
__device__ __forceinline__ float ord2f(unsigned k) {
  unsigned u = (k & 0x80000000u) ? (k ^ 0x80000000u) : ~k;
  return __uint_as_float(u);
}

// assemble a 16-half fragment from two contiguous 8-half vectors
__device__ __forceinline__ v16h frag16(const _Float16* lo, const _Float16* hs) {
  v8h a = *(const v8h*)lo;
  v8h b = *(const v8h*)hs;
  return __builtin_shufflevector(a, b, 0, 1, 2, 3, 4, 5, 6, 7,
                                       8, 9, 10, 11, 12, 13, 14, 15);
}

// ---- one-time prep: pack weights to fp16 in WMMA *fragment order* ----
__global__ void k_cvt(const float* pre_w, const float* post_w, const float* lin_w,
                      _Float16* preWp, _Float16* postWp, _Float16* linWp) {
  int i = blockIdx.x * blockDim.x + threadIdx.x;
  int stride = gridDim.x * blockDim.x;
  for (int p = i; p < 3 * PRE_NT * PRE_NC * 512; p += stride) {
    int l  = p / (PRE_NT * PRE_NC * 512);
    int r  = p % (PRE_NT * PRE_NC * 512);
    int nt = r / (PRE_NC * 512);
    int r2 = r % (PRE_NC * 512);
    int kc = r2 / 512;
    int r3 = r2 % 512;
    int lane = r3 / 16, j = r3 % 16;
    int n = lane & 15, hi = lane >> 4;
    int kb  = kc * 32 + j + hi * 16;
    int col = nt * 16 + n;
    float v = 0.f;
    if (kb < PRE_K && col < TD) {
      int t = col / DD, f = col % DD;
      v = pre_w[((size_t)(l * TT + t) * PRE_K + kb) * DD + f];
    }
    preWp[p] = (_Float16)v;
  }
  for (int p = i; p < 3 * TT * POST_NC * 512; p += stride) {
    int l  = p / (TT * POST_NC * 512);
    int r  = p % (TT * POST_NC * 512);
    int t  = r / (POST_NC * 512);
    int r2 = r % (POST_NC * 512);
    int kc = r2 / 512;
    int r3 = r2 % 512;
    int lane = r3 / 16, j = r3 % 16;
    int n = lane & 15, hi = lane >> 4;
    int kb = kc * 32 + j + hi * 16;
    float v = 0.f;
    if (kb < POST_K && n < FOUT_)
      v = post_w[((size_t)(l * TT + t) * POST_K + kb) * FOUT_ + n];
    postWp[p] = (_Float16)v;
  }
  for (int p = i; p < 3 * LIN_NT * LIN_NC * 512; p += stride) {
    int l  = p / (LIN_NT * LIN_NC * 512);
    int r  = p % (LIN_NT * LIN_NC * 512);
    int nt = r / (LIN_NC * 512);
    int r2 = r % (LIN_NC * 512);
    int kc = r2 / 512;
    int r3 = r2 % 512;
    int lane = r3 / 16, j = r3 % 16;
    int n = lane & 15, hi = lane >> 4;
    int kb  = kc * 32 + j + hi * 16;
    int col = nt * 16 + n;
    float v = 0.f;
    if (kb < DD && col < DD) v = lin_w[((size_t)l * DD + kb) * DD + col];
    linWp[p] = (_Float16)v;
  }
}

// ---- embeddings ----
__global__ void k_embed(const int* x, const int* eattr, const float* node_emb,
                        const float* edge_emb, float* h, _Float16* ea16) {
  int i = blockIdx.x * blockDim.x + threadIdx.x;
  int stride = gridDim.x * blockDim.x;
  for (int p = i; p < NNODES * DD; p += stride) {
    int n = p / DD, c = p % DD;
    h[p] = node_emb[x[n] * DD + c];
  }
  for (size_t p = i; p < (size_t)NEDGES * EDD; p += stride) {
    size_t e = p / EDD; int c = (int)(p % EDD);
    ea16[p] = (_Float16)edge_emb[eattr[e] * EDD + c];
  }
}

// ---- per-graph top-K via bitonic sort in LDS ----
__global__ __launch_bounds__(256) void k_topk(const float* score, int* sel) {
  int g = blockIdx.x, tid = threadIdx.x;
  __shared__ float v[NPG];
  __shared__ int   id[NPG];
  for (int i = tid; i < NPG; i += 256) { v[i] = score[g * NPG + i]; id[i] = i; }
  __syncthreads();
  for (int k = 2; k <= NPG; k <<= 1) {
    for (int j = k >> 1; j > 0; j >>= 1) {
      for (int i = tid; i < NPG; i += 256) {
        int ixj = i ^ j;
        if (ixj > i) {
          bool ddd = ((i & k) == 0);   // descending segment
          float va = v[i], vb = v[ixj];
          int ia = id[i], ib = id[ixj];
          bool gfirst = (va > vb) || (va == vb && ia < ib);
          if (gfirst != ddd) { v[i] = vb; v[ixj] = va; id[i] = ib; id[ixj] = ia; }
        }
      }
      __syncthreads();
    }
  }
  for (int i = tid; i < NPG; i += 256)
    sel[g * NPG + id[i]] = (i < TOPK) ? 1 : 0;
}

// ---- per-layer accumulator init (must run every call/layer) ----
__global__ void k_init(float* s1, float* s2, unsigned* mnb, unsigned* mxb,
                       float* deg, float* bnstat, int* cnt, float* dsum) {
  int i = blockIdx.x * blockDim.x + threadIdx.x;
  if (i < NNODES * TD) {
    s1[i] = 0.f; s2[i] = 0.f;
    mnb[i] = 0xFFFFFFFFu;   // max key
    mxb[i] = 0u;            // min key
  }
  if (i < NNODES) deg[i] = 0.f;
  if (i < 150)    bnstat[i] = 0.f;
  if (i == 0) { *cnt = 0; *dsum = 0.f; }
}

// ---- compact masked edges + in-degree ----
__global__ __launch_bounds__(256) void k_compact(const int* rowI, const int* colI,
                                                 const int* sel, int* kept, int* cnt,
                                                 float* deg) {
  int e = blockIdx.x * 256 + threadIdx.x;
  if (e < NEDGES && sel[rowI[e]]) {
    int p = atomicAdd(cnt, 1);
    kept[p] = e;
    atomicAdd(&deg[colI[e]], 1.f);
  }
}

// ---- delta = mean(log(deg+1)) ----
__global__ __launch_bounds__(256) void k_delta(const float* deg, float* dsum) {
  __shared__ float red[256];
  int n = blockIdx.x * 256 + threadIdx.x;
  red[threadIdx.x] = logf(deg[n] + 1.f);
  __syncthreads();
  for (int s = 128; s > 0; s >>= 1) {
    if (threadIdx.x < s) red[threadIdx.x] += red[threadIdx.x + s];
    __syncthreads();
  }
  if (threadIdx.x == 0) atomicAdd(dsum, red[0]);
}

// ---- pre-MLP messages on kept edges (WMMA) + fused segment reductions ----
__global__ __launch_bounds__(32) void k_premsg(
    const int* kept, const int* cntp, const int* rowI, const int* colI,
    const float* h, const _Float16* ea16, const v16h* Wp, const float* pre_bL,
    float* s1, float* s2, unsigned* mnb, unsigned* mxb) {
  int blk = blockIdx.x;
  int cnt = *cntp;
  if (blk * 16 >= cnt) return;                 // block-uniform exit
  int nrows = cnt - blk * 16; if (nrows > 16) nrows = 16;
  int ln = threadIdx.x;
  __shared__ __align__(16) _Float16 A[16][PRE_KP];
  __shared__ int eI[16], eD[16], eR[16];
  if (ln < 16) {
    int e = (ln < nrows) ? kept[blk * 16 + ln] : -1;
    eI[ln] = e;
    eD[ln] = (e >= 0) ? colI[e] : 0;
    eR[ln] = (e >= 0) ? rowI[e] : 0;
  }
  __syncthreads();
  for (int p = ln; p < 16 * PRE_KP; p += 32) {
    int m = p / PRE_KP, c = p % PRE_KP;
    float v = 0.f;
    int e = eI[m];
    if (e >= 0) {
      if (c < DD)            v = h[(size_t)eD[m] * DD + c];          // x_i = h[col]
      else if (c < 2 * DD)   v = h[(size_t)eR[m] * DD + (c - DD)];   // x_j = h[row]
      else if (c < PRE_K)    v = (float)ea16[(size_t)e * EDD + (c - 2 * DD)];
    }
    A[m][c] = (_Float16)v;
  }
  __syncthreads();
  int n = ln & 15, hi = ln >> 4;
  for (int nt = 0; nt < PRE_NT; ++nt) {
    v8f acc = {0.f, 0.f, 0.f, 0.f, 0.f, 0.f, 0.f, 0.f};
    #pragma unroll
    for (int kc = 0; kc < PRE_NC; ++kc) {
      v16h a = frag16(&A[n][kc * 32 + hi * 16], &A[n][kc * 32 + hi * 16 + 8]);
      v16h b = Wp[(nt * PRE_NC + kc) * 32 + ln];
      acc = __builtin_amdgcn_wmma_f32_16x16x32_f16(false, a, false, b,
                                                   (short)0, acc, false, false);
    }
    int gcol = nt * 16 + n;
    #pragma unroll
    for (int i = 0; i < 8; ++i) {
      int m = i + 8 * hi;
      if (m < nrows && gcol < TD) {
        float v = acc[i] + pre_bL[gcol];
        size_t o = (size_t)eD[m] * TD + gcol;
        atomicAdd(s1 + o, v);
        atomicAdd(s2 + o, v * v);
        unsigned key = f2ord(v);
        atomicMax(mxb + o, key);
        atomicMin(mnb + o, key);
      }
    }
  }
}

// ---- post-MLP: build [16,975] tile on the fly, GEMM vs post_w[t] (WMMA) ----
__global__ __launch_bounds__(32) void k_post(
    const float* h, const float* deg, const float* dsum,
    const float* s1, const float* s2, const unsigned* mnb, const unsigned* mxb,
    const v16h* Wp, const float* post_bL, float* out75) {
  int tile = blockIdx.x, t = blockIdx.y;
  int ln = threadIdx.x;
  __shared__ __align__(16) _Float16 A[16][POST_KP];
  __shared__ float rAmp[16], rAtt[16], rCnt[16], rDeg[16];
  float delta = dsum[0] * (1.f / NNODES);
  if (ln < 16) {
    int node = tile * 16 + ln;
    float d = deg[node];
    float cv = fmaxf(d, 1.f);
    float ld = logf(cv + 1.f);
    rDeg[ln] = d; rCnt[ln] = cv;
    rAmp[ln] = ld / delta; rAtt[ln] = delta / ld;
  }
  __syncthreads();
  for (int p = ln; p < 16 * POST_KP; p += 32) {
    int m = p / POST_KP, c = p % POST_KP;
    int node = tile * 16 + m;
    float v = 0.f;
    if (c < DD) v = h[(size_t)node * DD + c];
    else if (c < POST_K) {
      int c2 = c - DD;
      int sblk = c2 / 300;
      int within = c2 % 300;
      int a = within / DD, f = within % DD;
      size_t o = (size_t)node * TD + t * DD + f;
      float cv = rCnt[m], base;
      if (a == 0)      base = s1[o] / cv;
      else if (a == 1) base = (rDeg[m] > 0.f) ? ord2f(mnb[o]) : 0.f;
      else if (a == 2) base = (rDeg[m] > 0.f) ? ord2f(mxb[o]) : 0.f;
      else {
        float mean = s1[o] / cv;
        float var  = s2[o] / cv - mean * mean;
        base = sqrtf(fmaxf(var, 0.f) + EPS_);
      }
      float sc = (sblk == 0) ? 1.f : ((sblk == 1) ? rAmp[m] : rAtt[m]);
      v = base * sc;
    }
    A[m][c] = (_Float16)v;
  }
  __syncthreads();
  int n = ln & 15, hi = ln >> 4;
  const v16h* Wt = Wp + (size_t)t * POST_NC * 32;   // tower offset
  v8f acc = {0.f, 0.f, 0.f, 0.f, 0.f, 0.f, 0.f, 0.f};
  for (int kc = 0; kc < POST_NC; ++kc) {
    v16h a = frag16(&A[n][kc * 32 + hi * 16], &A[n][kc * 32 + hi * 16 + 8]);
    v16h b = Wt[kc * 32 + ln];
    acc = __builtin_amdgcn_wmma_f32_16x16x32_f16(false, a, false, b,
                                                 (short)0, acc, false, false);
  }
  #pragma unroll
  for (int i = 0; i < 8; ++i) {
    int m = i + 8 * hi; int node = tile * 16 + m;
    if (n < FOUT_)
      out75[(size_t)node * DD + t * FOUT_ + n] = acc[i] + post_bL[t * FOUT_ + n];
  }
}

// ---- final linear 75x75 (WMMA) + BN partial sums ----
__global__ __launch_bounds__(32) void k_lin(
    const float* out75, const v16h* Wp, const float* lin_bL,
    float* lin_out, float* bnstat) {
  int tile = blockIdx.x, nt = blockIdx.y, ln = threadIdx.x;
  __shared__ __align__(16) _Float16 A[16][LIN_KP];
  for (int p = ln; p < 16 * LIN_KP; p += 32) {
    int m = p / LIN_KP, c = p % LIN_KP;
    float v = (c < DD) ? out75[(size_t)(tile * 16 + m) * DD + c] : 0.f;
    A[m][c] = (_Float16)v;
  }
  __syncthreads();
  int n = ln & 15, hi = ln >> 4;
  v8f acc = {0.f, 0.f, 0.f, 0.f, 0.f, 0.f, 0.f, 0.f};
  #pragma unroll
  for (int kc = 0; kc < LIN_NC; ++kc) {
    v16h a = frag16(&A[n][kc * 32 + hi * 16], &A[n][kc * 32 + hi * 16 + 8]);
    v16h b = Wp[(nt * LIN_NC + kc) * 32 + ln];
    acc = __builtin_amdgcn_wmma_f32_16x16x32_f16(false, a, false, b,
                                                 (short)0, acc, false, false);
  }
  int gcol = nt * 16 + n;
  float ps = 0.f, pq = 0.f;
  #pragma unroll
  for (int i = 0; i < 8; ++i) {
    int m = i + 8 * hi; int node = tile * 16 + m;
    if (gcol < DD) {
      float v = acc[i] + lin_bL[gcol];
      lin_out[(size_t)node * DD + gcol] = v;
      ps += v; pq += v * v;
    }
  }
  if (gcol < DD) {
    atomicAdd(&bnstat[gcol], ps);
    atomicAdd(&bnstat[75 + gcol], pq);
  }
}

// ---- BatchNorm apply + score MLP (75->50->25->1) ----
__global__ __launch_bounds__(256) void k_bn_mlp(
    const float* lin_out, const float* bnstat, const float* bn_g, const float* bn_b,
    const float* rel, const float* w1, const float* b1, const float* w2,
    const float* b2, const float* w3, const float* b3,
    float* h, float* score_out) {
  __shared__ float sw1[DD * 50];
  __shared__ float sw2[50 * 25];
  __shared__ float sw3[25], sb1[50], sb2[25];
  __shared__ float smu[DD], srs[DD], sg[DD], sbb[DD];
  int tid = threadIdx.x;
  for (int i = tid; i < DD * 50; i += 256) sw1[i] = w1[i];
  for (int i = tid; i < 50 * 25; i += 256) sw2[i] = w2[i];
  if (tid < 25) { sw3[tid] = w3[tid]; sb2[tid] = b2[tid]; }
  if (tid < 50) sb1[tid] = b1[tid];
  if (tid < DD) {
    float mu  = bnstat[tid] * (1.f / NNODES);
    float var = bnstat[75 + tid] * (1.f / NNODES) - mu * mu;
    smu[tid] = mu; srs[tid] = rsqrtf(var + EPS_);
    sg[tid] = bn_g[tid]; sbb[tid] = bn_b[tid];
  }
  __syncthreads();
  int node = blockIdx.x * 256 + tid;
  float a1[50];
  #pragma unroll
  for (int j = 0; j < 50; ++j) a1[j] = sb1[j];
  for (int c = 0; c < DD; ++c) {
    float v = (lin_out[(size_t)node * DD + c] - smu[c]) * srs[c] * sg[c] + sbb[c];
    h[(size_t)node * DD + c] = v;
    float hr = v + rel[(size_t)node * DD + c];
    #pragma unroll
    for (int j = 0; j < 50; ++j) a1[j] += hr * sw1[c * 50 + j];
  }
  float a2[25];
  #pragma unroll
  for (int k = 0; k < 25; ++k) a2[k] = sb2[k];
  #pragma unroll
  for (int j = 0; j < 50; ++j) {
    float aj = fmaxf(a1[j], 0.f);
    #pragma unroll
    for (int k = 0; k < 25; ++k) a2[k] += aj * sw2[j * 25 + k];
  }
  float s = b3[0];
  #pragma unroll
  for (int k = 0; k < 25; ++k) s += fmaxf(a2[k], 0.f) * sw3[k];
  score_out[node] = s;
}

extern "C" void kernel_launch(void* const* d_in, const int* in_sizes, int n_in,
                              void* d_out, int out_size, void* d_ws, size_t ws_size,
                              hipStream_t stream) {
  (void)in_sizes; (void)n_in; (void)out_size; (void)ws_size;
  const int*   x        = (const int*)d_in[0];
  const int*   eidx     = (const int*)d_in[1];
  const int*   eattr    = (const int*)d_in[2];
  /* d_in[3] = batch (unused: layout is graph-major by construction) */
  const float* score0   = (const float*)d_in[4];
  const float* rel      = (const float*)d_in[5];
  const float* node_emb = (const float*)d_in[6];
  const float* edge_emb = (const float*)d_in[7];
  const float* pre_w    = (const float*)d_in[8];
  const float* pre_b    = (const float*)d_in[9];
  const float* post_w   = (const float*)d_in[10];
  const float* post_b   = (const float*)d_in[11];
  const float* lin_w    = (const float*)d_in[12];
  const float* lin_b    = (const float*)d_in[13];
  const float* bn_g     = (const float*)d_in[14];
  const float* bn_b     = (const float*)d_in[15];
  const float* w1 = (const float*)d_in[16];
  const float* b1 = (const float*)d_in[17];
  const float* w2 = (const float*)d_in[18];
  const float* b2 = (const float*)d_in[19];
  const float* w3 = (const float*)d_in[20];
  const float* b3 = (const float*)d_in[21];
  const int* rowI = eidx;            // edge_index[0] = src
  const int* colI = eidx + NEDGES;   // edge_index[1] = dst

  size_t off = 0;
  auto alloc = [&](size_t bytes) -> void* {
    void* p = (char*)d_ws + off;
    off += (bytes + 255) & ~(size_t)255;
    return p;
  };
  float*    h      = (float*)   alloc(sizeof(float) * NNODES * DD);
  _Float16* ea16   = (_Float16*)alloc(sizeof(_Float16) * (size_t)NEDGES * EDD);
  _Float16* preWp  = (_Float16*)alloc(sizeof(_Float16) * 3 * PRE_NT * PRE_NC * 512);
  _Float16* postWp = (_Float16*)alloc(sizeof(_Float16) * 3 * TT * POST_NC * 512);
  _Float16* linWp  = (_Float16*)alloc(sizeof(_Float16) * 3 * LIN_NT * LIN_NC * 512);
  int*      sel  = (int*)     alloc(sizeof(int) * NNODES);
  float*    deg  = (float*)   alloc(sizeof(float) * NNODES);
  int*      kept = (int*)     alloc(sizeof(int) * NEDGES);
  int*      cnt  = (int*)     alloc(256);
  float*    dsum = (float*)   alloc(256);
  float*    s1   = (float*)   alloc(sizeof(float) * (size_t)NNODES * TD);
  float*    s2   = (float*)   alloc(sizeof(float) * (size_t)NNODES * TD);
  unsigned* mnb  = (unsigned*)alloc(sizeof(unsigned) * (size_t)NNODES * TD);
  unsigned* mxb  = (unsigned*)alloc(sizeof(unsigned) * (size_t)NNODES * TD);
  float* out75   = (float*)alloc(sizeof(float) * NNODES * DD);
  float* lin_out = (float*)alloc(sizeof(float) * NNODES * DD);
  float* bnstat  = (float*)alloc(1024);
  float* score_buf = (float*)alloc(sizeof(float) * NNODES);

  k_cvt  <<<512,  256, 0, stream>>>(pre_w, post_w, lin_w, preWp, postWp, linWp);
  k_embed<<<2048, 256, 0, stream>>>(x, eattr, node_emb, edge_emb, h, ea16);

  for (int l = 0; l < 3; ++l) {
    const float* ssrc = (l == 0) ? score0 : score_buf;
    float*       sdst = (l == 2) ? (float*)d_out : score_buf;
    k_topk   <<<BGRAPH, 256, 0, stream>>>(ssrc, sel);
    k_init   <<<(NNODES * TD + 255) / 256, 256, 0, stream>>>(s1, s2, mnb, mxb, deg,
                                                             bnstat, cnt, dsum);
    k_compact<<<NEDGES / 256, 256, 0, stream>>>(rowI, colI, sel, kept, cnt, deg);
    k_delta  <<<NNODES / 256, 256, 0, stream>>>(deg, dsum);
    k_premsg <<<NEDGES / 16, 32, 0, stream>>>(
        kept, cnt, rowI, colI, h, ea16,
        (const v16h*)(preWp + (size_t)l * PRE_NT * PRE_NC * 512),
        pre_b + l * TD, s1, s2, mnb, mxb);
    k_post   <<<dim3(NNODES / 16, TT), 32, 0, stream>>>(
        h, deg, dsum, s1, s2, mnb, mxb,
        (const v16h*)(postWp + (size_t)l * TT * POST_NC * 512),
        post_b + l * TT * FOUT_, out75);
    k_lin    <<<dim3(NNODES / 16, LIN_NT), 32, 0, stream>>>(
        out75, (const v16h*)(linWp + (size_t)l * LIN_NT * LIN_NC * 512),
        lin_b + l * DD, lin_out, bnstat);
    k_bn_mlp <<<NNODES / 256, 256, 0, stream>>>(lin_out, bnstat, bn_g + l * DD,
                                              bn_b + l * DD, rel, w1, b1, w2, b2,
                                              w3, b3, h, sdst);
  }
}
